// RUM_72980084294375
// MI455X (gfx1250) — matrix operand order repaired and checked by reading, not compiled, not run on hardware
//
#include <hip/hip_runtime.h>
#include <hip/hip_bf16.h>

typedef __attribute__((ext_vector_type(16))) _Float16 v16h;
typedef __attribute__((ext_vector_type(8)))  float    v8f;
typedef __attribute__((ext_vector_type(4)))  float    v4f;

#define RUM_L 50
#define RUM_E 64

// One wave (32 lanes) per batch element.
// w = X_e @ y_e via v_wmma_f32_16x16x32_f16:
//   M = history slot l (4 tiles of 16, l=50..63 padded with zero row),
//   K = embedding dim  (2 steps of 32),
//   N = 16 replicated copies of y  -> every D column equals w.
// out = sigmoid( sum_l w_l softmax(w)_l + dot(u_e, y_e) )   [alpha=beta=1]
__global__ __launch_bounds__(256) void rum_wmma_kernel(
    const int*   __restrict__ u,
    const int*   __restrict__ X,
    const int*   __restrict__ y,
    const float* __restrict__ item_emb,
    const float* __restrict__ user_emb,
    float*       __restrict__ out,
    int B)
{
    const int lane = threadIdx.x & 31;
    const int wave = threadIdx.x >> 5;
    const int b    = blockIdx.x * 8 + wave;
    if (b >= B) return;                       // wave-uniform: EXEC stays all-ones for WMMA

    const bool hi     = lane >= 16;
    const int  m      = lane & 15;
    const int  koff_a = hi ? 8 : 0;           // A 16-bit 16x32 layout: hi half covers K+8
    const int  koff_b = hi ? 8 : 0;           // B 16-bit 32x16 layout: hi half covers K+8

    const float* yrow = item_emb + (size_t)y[b] * RUM_E;

    // ---- B operand: y broadcast to all 16 N columns; per lane two 8-float runs
    //      at +0 and +16 => 4x b128 broadcast loads + cvt to f16.
    v16h bmat[2];
#pragma unroll
    for (int ks = 0; ks < 2; ++ks) {
        const int base = ks * 32 + koff_b;
        v4f f4[4];
        f4[0] = *(const v4f*)(yrow + base + 0);
        f4[1] = *(const v4f*)(yrow + base + 4);
        f4[2] = *(const v4f*)(yrow + base + 16);
        f4[3] = *(const v4f*)(yrow + base + 20);
        const float* f = (const float*)f4;
        v16h bb;
#pragma unroll
        for (int i = 0; i < 16; ++i) bb[i] = (_Float16)f[i];
        bmat[ks] = bb;
    }

    // ---- accumulators: 4 l-tiles of 16 rows
    v8f acc[4];
#pragma unroll
    for (int t = 0; t < 4; ++t)
#pragma unroll
        for (int r = 0; r < 8; ++r) acc[t][r] = 0.0f;

#pragma unroll
    for (int t = 0; t < 4; ++t) {
        const int l  = t * 16 + m;                    // lanes m and m+16 share row l
        const int lc = (l < RUM_L) ? l : 0;
        int idx = X[(size_t)b * RUM_L + lc];
        if (l >= RUM_L) idx = 0;                      // item 0 = all-zero padding row
        const float* arow = item_emb + (size_t)idx * RUM_E;
#pragma unroll
        for (int ks = 0; ks < 2; ++ks) {
            // lane needs runs [e0+koff_a, +8) and [e0+16+koff_a, +8)
            const int r0 = ks * 32 + koff_a;
            v4f f4[4];
            f4[0] = *(const v4f*)(arow + r0 + 0);
            f4[1] = *(const v4f*)(arow + r0 + 4);
            f4[2] = *(const v4f*)(arow + r0 + 16);
            f4[3] = *(const v4f*)(arow + r0 + 20);
            const float* f = (const float*)f4;
            v16h a;
#pragma unroll
            for (int i = 0; i < 16; ++i) a[i] = (_Float16)f[i];
            acc[t] = __builtin_amdgcn_wmma_f32_16x16x32_f16(
                /*neg_a=*/false, a, /*neg_b=*/false, bmat[ks],
                /*c_mod=*/(short)0, acc[t], /*reuse_a=*/false, /*reuse_b=*/false);
        }
    }

    // acc[t][r] == w[t*16 + r + (hi?8:0)], identical in every lane of its half.
    const int hioff = hi ? 8 : 0;
    float mymax = -3.4e38f;
#pragma unroll
    for (int t = 0; t < 4; ++t)
#pragma unroll
        for (int r = 0; r < 8; ++r) {
            const int l = t * 16 + hioff + r;
            mymax = fmaxf(mymax, (l < RUM_L) ? acc[t][r] : -3.4e38f);
        }
    mymax = fmaxf(mymax, __shfl_xor(mymax, 16));

    float ssum = 0.0f, tsum = 0.0f;               // sum e^w, sum w e^w
#pragma unroll
    for (int t = 0; t < 4; ++t)
#pragma unroll
        for (int r = 0; r < 8; ++r) {
            const int   l = t * 16 + hioff + r;
            const float w = acc[t][r];            // finite even when padded (zero row)
            const float e = (l < RUM_L) ? __expf(w - mymax) : 0.0f;
            ssum += e;
            tsum += w * e;
        }
    ssum += __shfl_xor(ssum, 16);
    tsum += __shfl_xor(tsum, 16);

    // dot(u_e, y_e): float2 per lane, 5-step butterfly
    const float* urow = user_emb + (size_t)u[b] * RUM_E;
    float2 uu = *(const float2*)(urow + 2 * lane);
    float2 yy = *(const float2*)(yrow + 2 * lane);
    float d = uu.x * yy.x + uu.y * yy.y;
#pragma unroll
    for (int off = 16; off >= 1; off >>= 1) d += __shfl_xor(d, off);

    const float logit = tsum / ssum + d;
    if (lane == 0) out[b] = 1.0f / (1.0f + __expf(-logit));
}

extern "C" void kernel_launch(void* const* d_in, const int* in_sizes, int n_in,
                              void* d_out, int out_size, void* d_ws, size_t ws_size,
                              hipStream_t stream) {
    const int*   u        = (const int*)d_in[0];
    const int*   X        = (const int*)d_in[1];
    const int*   y        = (const int*)d_in[2];
    const float* item_emb = (const float*)d_in[3];
    const float* user_emb = (const float*)d_in[4];
    float*       out      = (float*)d_out;
    const int B = in_sizes[0];

    dim3 block(256);                 // 8 wave32 waves -> 8 batch elements per block
    dim3 grid((B + 7) / 8);
    rum_wmma_kernel<<<grid, block, 0, stream>>>(u, X, y, item_emb, user_emb, out, B);
}
